// GCNEncoder_31593779429472
// MI455X (gfx1250) — compile-verified, matrix operand order checked by reading, not run on hardware
//
#include <hip/hip_runtime.h>
#include <hip/hip_bf16.h>

#define NFEAT_IN 500
#define NHID 128
#define NOUT 64

typedef __attribute__((ext_vector_type(16))) __bf16    v16bf;
typedef __attribute__((ext_vector_type(8)))  float     v8f;
typedef __attribute__((ext_vector_type(4)))  unsigned  u32x4;
typedef __attribute__((ext_vector_type(2)))  unsigned  u32x2;
typedef __attribute__((ext_vector_type(8)))  unsigned  u32x8;
typedef __attribute__((ext_vector_type(4)))  float     f32x4;
typedef __attribute__((ext_vector_type(2)))  float     f32x2;

// round-to-nearest-even f32 -> bf16, packed pair into one dword (lo | hi<<16)
static __device__ __forceinline__ unsigned bfpack(float lo, float hi) {
    unsigned ul = __builtin_bit_cast(unsigned, lo);
    unsigned uh = __builtin_bit_cast(unsigned, hi);
    unsigned rl = (ul + 0x7FFFu + ((ul >> 16) & 1u)) >> 16;
    unsigned rh = (uh + 0x7FFFu + ((uh >> 16) & 1u)) & 0xFFFF0000u;
    return (rl & 0xFFFFu) | rh;
}

// ---------------------------------------------------------------------------
// Degree / normalization
// ---------------------------------------------------------------------------
__global__ void k_deg_init(float* __restrict__ deg, int n) {
    int i = blockIdx.x * blockDim.x + threadIdx.x;
    if (i < n) deg[i] = 1.0f;                    // self-loop contributes 1
}

__global__ void k_deg_edges(const int* __restrict__ dst, float* __restrict__ deg, int e) {
    int i = blockIdx.x * blockDim.x + threadIdx.x;
    if (i < e) unsafeAtomicAdd(&deg[dst[i]], 1.0f);
}

__global__ void k_deg_to_dis(float* __restrict__ deg, int n) {
    int i = blockIdx.x * blockDim.x + threadIdx.x;
    if (i < n) deg[i] = rsqrtf(deg[i]);          // deg >= 1 always (self loops)
}

// ---------------------------------------------------------------------------
// WMMA GEMM: C[nrows x NDIM] = A[nrows x KDIM] * B[KDIM x NDIM]
// fp32 in/out, bf16 multiply via v_wmma_f32_16x16x32_bf16 (wave32).
// Block: 16 rows, NDIM/16 waves (one 16x16 tile per wave).
// LDS holds bf16 K-pairs packed in dwords:
//   AsPk[m][q]  : A row m, pair q covers K = 2q,2q+1   (20-dword padded rows)
//   BsPk[n][q]  : B col n, pair q covers K = 2q,2q+1   (20-dword padded cols)
// Operand fetch per WMMA = two 16B ds_load_b128 per matrix.
// Main K loop is branch-free; the partial K chunk (if any) is peeled.
// Requires nrows % 16 == 0 (true here: 50000).
// ---------------------------------------------------------------------------
template<int KDIM, int NDIM>
__global__ void k_gemm_wmma(const float* __restrict__ A, const float* __restrict__ B,
                            float* __restrict__ C, int nrows) {
    constexpr int NT   = NDIM / 16;              // waves per block
    constexpr int PAD  = 20;                     // dwords per 16-pair row (80B, 16B-aligned)
    __shared__ unsigned AsPk[16 * PAD];          // 1.25 KB
    __shared__ unsigned BsPk[NDIM * PAD];        // 10 KB (NDIM=128) / 5 KB (64)

    const int tid  = threadIdx.x;
    const int wave = tid >> 5;
    const int lane = tid & 31;
    const int half = lane >> 4;                  // 0: lanes 0-15, 1: lanes 16-31
    const int l16  = lane & 15;
    const int row0 = blockIdx.x * 16;
    const int col0 = wave * 16;

    v8f acc = {};

    // stage one 32-wide K chunk into LDS (TAIL: per-element bounds checks)
    auto stage = [&](int k0, auto tail_tag) {
        constexpr bool TAIL = decltype(tail_tag)::value;
        // A tile: 16 rows x 32 K -> 128 float4 loads (A rows are 16B-aligned)
        for (int i = tid; i < 128; i += NT * 32) {
            int r = i >> 3, seg = i & 7;
            int gk = k0 + seg * 4;
            const float* ap = A + (size_t)(row0 + r) * KDIM + gk;
            f32x4 v;
            if constexpr (TAIL) {
                v[0] = (gk + 0 < KDIM) ? ap[0] : 0.0f;
                v[1] = (gk + 1 < KDIM) ? ap[1] : 0.0f;
                v[2] = (gk + 2 < KDIM) ? ap[2] : 0.0f;
                v[3] = (gk + 3 < KDIM) ? ap[3] : 0.0f;
            } else {
                v = *(const f32x4*)ap;           // unconditional global_load_b128
            }
            u32x2 pk = { bfpack(v[0], v[1]), bfpack(v[2], v[3]) };
            *(u32x2*)(&AsPk[r * PAD + seg * 2]) = pk;
        }
        // B slab: 32 K x NDIM, transposed into packed pairs per column
        for (int i = tid; i < 16 * NDIM; i += NT * 32) {
            int n = i & (NDIM - 1), p = i / NDIM;   // pair p covers k0+2p, k0+2p+1
            int gk = k0 + 2 * p;
            float f0, f1;
            if constexpr (TAIL) {
                f0 = (gk     < KDIM) ? B[(size_t)gk * NDIM + n]       : 0.0f;
                f1 = (gk + 1 < KDIM) ? B[(size_t)(gk + 1) * NDIM + n] : 0.0f;
            } else {
                f0 = B[(size_t)gk * NDIM + n];
                f1 = B[(size_t)(gk + 1) * NDIM + n];
            }
            BsPk[n * PAD + p] = bfpack(f0, f1);
        }
    };

    // operand gather (pair index q = (v<4?0:8) + half*4 + (v&3)) + one WMMA
    auto mma_step = [&]() {
        const unsigned* ab = &AsPk[l16 * PAD];
        const unsigned* bb = &BsPk[(col0 + l16) * PAD];
        u32x4 alo = *(const u32x4*)(ab + half * 4);        // ds_load_b128
        u32x4 ahi = *(const u32x4*)(ab + 8 + half * 4);
        u32x4 blo = *(const u32x4*)(bb + half * 4);
        u32x4 bhi = *(const u32x4*)(bb + 8 + half * 4);
        u32x8 av = { alo[0], alo[1], alo[2], alo[3], ahi[0], ahi[1], ahi[2], ahi[3] };
        u32x8 bv = { blo[0], blo[1], blo[2], blo[3], bhi[0], bhi[1], bhi[2], bhi[3] };
        v16bf a = __builtin_bit_cast(v16bf, av);
        v16bf b = __builtin_bit_cast(v16bf, bv);
        acc = __builtin_amdgcn_wmma_f32_16x16x32_bf16(
            /*neg_a=*/false, a, /*neg_b=*/false, b,
            /*c_mod=*/(short)0, acc, /*reuse_a=*/false, /*reuse_b=*/false);
    };

    constexpr int KFULL = (KDIM / 32) * 32;
    for (int k0 = 0; k0 < KFULL; k0 += 32) {     // branch-free hot loop
        stage(k0, std::integral_constant<bool, false>{});
        __syncthreads();
        mma_step();
        __syncthreads();
    }
    if constexpr (KDIM % 32 != 0) {              // peeled partial chunk
        stage(KFULL, std::integral_constant<bool, true>{});
        __syncthreads();
        mma_step();
        __syncthreads();
    }

    // C/D layout: VGPR r -> M = r + 8*half, N = lane&15
#pragma unroll
    for (int r = 0; r < 8; ++r) {
        int gr = row0 + half * 8 + r;
        C[(size_t)gr * NDIM + col0 + l16] = acc[r];
    }
}

// ---------------------------------------------------------------------------
// Aggregation
// ---------------------------------------------------------------------------

// agg[i,:] = dis[i]^2 * t[i,:] (+ bias)  : self-loop contribution, full overwrite
template<int F, bool ADD_BIAS>
__global__ void k_self_init(const float* __restrict__ dis, const float* __restrict__ t,
                            const float* __restrict__ bias, float* __restrict__ agg, int n) {
    long long idx = (long long)blockIdx.x * blockDim.x + threadIdx.x;
    if (idx >= (long long)n * F) return;
    int i = (int)(idx / F);
    int f = (int)(idx % F);
    float d = dis[i];
    float v = d * d * t[idx];
    if (ADD_BIAS) v += bias[f];
    agg[idx] = v;
}

// one wave32 per edge: agg[dst,:] += dis[src]*dis[dst] * t[src,:]
template<int F>
__global__ void k_edge_scatter(const int* __restrict__ src, const int* __restrict__ dst,
                               const float* __restrict__ dis, const float* __restrict__ t,
                               float* __restrict__ agg, int e) {
    int wave = (int)((blockIdx.x * (long long)blockDim.x + threadIdx.x) >> 5);
    int lane = threadIdx.x & 31;
    if (wave >= e) return;
    int s = src[wave], d = dst[wave];
    float w = dis[s] * dis[d];
    constexpr int V = F / 32;                      // 4 floats (F=128) / 2 (F=64)
    const float* ts = t + (size_t)s * F + lane * V;
    float*       ad = agg + (size_t)d * F + lane * V;
    if constexpr (V == 4) {
        f32x4 v = *(const f32x4*)ts;               // global_load_b128
        unsafeAtomicAdd(ad + 0, w * v[0]);
        unsafeAtomicAdd(ad + 1, w * v[1]);
        unsafeAtomicAdd(ad + 2, w * v[2]);
        unsafeAtomicAdd(ad + 3, w * v[3]);
    } else {
        f32x2 v = *(const f32x2*)ts;               // global_load_b64
        unsafeAtomicAdd(ad + 0, w * v[0]);
        unsafeAtomicAdd(ad + 1, w * v[1]);
    }
}

// h[i,f] = relu(agg[i,f] + b[f])  (in place)
template<int F>
__global__ void k_relu_bias(float* __restrict__ agg, const float* __restrict__ bias, int n) {
    long long idx = (long long)blockIdx.x * blockDim.x + threadIdx.x;
    if (idx >= (long long)n * F) return;
    int f = (int)(idx % F);
    float v = agg[idx] + bias[f];
    agg[idx] = v > 0.0f ? v : 0.0f;
}

// ---------------------------------------------------------------------------
extern "C" void kernel_launch(void* const* d_in, const int* in_sizes, int n_in,
                              void* d_out, int out_size, void* d_ws, size_t ws_size,
                              hipStream_t stream) {
    const float* x    = (const float*)d_in[0];
    const int*   eidx = (const int*)d_in[1];
    const float* W1   = (const float*)d_in[2];
    const float* b1   = (const float*)d_in[3];
    const float* W2   = (const float*)d_in[4];
    const float* b2   = (const float*)d_in[5];
    float* out = (float*)d_out;

    const int N = in_sizes[0] / NFEAT_IN;          // 50000
    const int E = in_sizes[1] / 2;                 // 800000
    const int* src = eidx;                         // edge_index[0]
    const int* dst = eidx + E;                     // edge_index[1]

    // workspace layout (floats)
    float* wsf  = (float*)d_ws;
    float* dis  = wsf;                             // N
    float* t1   = dis + ((N + 63) & ~63);          // N * NHID
    float* agg1 = t1 + (size_t)N * NHID;           // N * NHID  (becomes h after relu)
    float* t2   = agg1 + (size_t)N * NHID;         // N * NOUT

    const int TB = 256;

    // ---- degree / normalization -------------------------------------------
    k_deg_init   <<<(N + TB - 1) / TB, TB, 0, stream>>>(dis, N);
    k_deg_edges  <<<(E + TB - 1) / TB, TB, 0, stream>>>(dst, dis, E);
    k_deg_to_dis <<<(N + TB - 1) / TB, TB, 0, stream>>>(dis, N);

    // ---- layer 1 -----------------------------------------------------------
    k_gemm_wmma<NFEAT_IN, NHID><<<(N + 15) / 16, (NHID / 16) * 32, 0, stream>>>(x, W1, t1, N);
    k_self_init<NHID, false><<<(((long long)N * NHID + TB - 1) / TB), TB, 0, stream>>>(dis, t1, b1, agg1, N);
    k_edge_scatter<NHID><<<(E + (TB / 32) - 1) / (TB / 32), TB, 0, stream>>>(src, dst, dis, t1, agg1, E);
    k_relu_bias<NHID><<<(((long long)N * NHID + TB - 1) / TB), TB, 0, stream>>>(agg1, b1, N);

    // ---- layer 2 -----------------------------------------------------------
    k_gemm_wmma<NHID, NOUT><<<(N + 15) / 16, (NOUT / 16) * 32, 0, stream>>>(agg1, W2, t2, N);
    k_self_init<NOUT, true><<<(((long long)N * NOUT + TB - 1) / TB), TB, 0, stream>>>(dis, t2, b2, out, N);
    k_edge_scatter<NOUT><<<(E + (TB / 32) - 1) / (TB / 32), TB, 0, stream>>>(src, dst, dis, t2, out, E);
}
